// LightweightGravitationalBlock_32993938768209
// MI455X (gfx1250) — matrix operand order repaired and checked by reading, not compiled, not run on hardware
//
#include <hip/hip_runtime.h>

// ---- problem constants (match reference) ----
#define B_  4
#define S_  2048
#define D_  256
#define P_  64
#define H_  8
#define FF_ 512
#define DH_ 32          // D_/H_
#define FMAX_ 50.0f
#define LNEPS_ 1e-5f

typedef __bf16 bf16_t;
typedef bf16_t v16bf __attribute__((ext_vector_type(16)));
typedef float  v8f   __attribute__((ext_vector_type(8)));
typedef unsigned short v16u __attribute__((ext_vector_type(16)));
typedef unsigned short v8u  __attribute__((ext_vector_type(8)));

#if defined(__has_builtin)
#if __has_builtin(__builtin_amdgcn_tensor_load_to_lds) && __has_builtin(__builtin_amdgcn_s_wait_tensorcnt)
#define HAVE_TDM 1
#endif
#endif
#ifndef HAVE_TDM
#define HAVE_TDM 0
#endif

__device__ __forceinline__ unsigned short f2bf(float f) {
  union { float f; unsigned int u; } v; v.f = f;
  unsigned int u = v.u;
  return (unsigned short)((u + 0x7FFFu + ((u >> 16) & 1u)) >> 16); // RNE
}
__device__ __forceinline__ v8f wmma_bf16(v16u a, v16u b, v8f c) {
  return __builtin_amdgcn_wmma_f32_16x16x32_bf16(
      false, __builtin_bit_cast(v16bf, a),
      false, __builtin_bit_cast(v16bf, b),
      (short)0, c, false, false);
}
// A-fragment: lane=M row; element K runs are [8h..8h+8) and [16+8h..24+8h) -> 2x b128
__device__ __forceinline__ v16u load_afrag(const unsigned short* p, int half) {
  union { v16u v; v8u h[2]; } u;
  u.h[0] = *(const v8u*)(p + 8 * half);
  u.h[1] = *(const v8u*)(p + 16 + 8 * half);
  return u.v;
}
// B-fragment: lane=N col; element K runs are [8h..8h+8) and [16+8h..24+8h) -> 2x b128
__device__ __forceinline__ v16u load_bfrag(const unsigned short* p, int half) {
  union { v16u v; v8u h[2]; } u;
  u.h[0] = *(const v8u*)(p + 8 * half);
  u.h[1] = *(const v8u*)(p + 16 + 8 * half);
  return u.v;
}

#if HAVE_TDM
typedef unsigned int u32x4 __attribute__((ext_vector_type(4)));
typedef int i32x4 __attribute__((ext_vector_type(4)));
typedef int i32x8 __attribute__((ext_vector_type(8)));
// 2D tile DMA: global[tileY rows x tileX elems, row stride strideElems] -> LDS (packed)
// D# per CDNA5 ISA 8.3/8.4: count=1, type=2(image), dataSizeCode: 0=1B,1=2B,2=4B.
__device__ __forceinline__ void tdm_load_2d(unsigned ldsOff, unsigned long long gaddr,
                                            int dataSizeCode, int tileX, int tileY,
                                            unsigned long long strideElems,
                                            unsigned tensorDim0) {
  u32x4 g0;
  g0.x = 1u;                                                  // count=1, user mode
  g0.y = ldsOff;                                              // lds_addr
  g0.z = (unsigned)(gaddr & 0xFFFFFFFFull);                   // global_addr lo
  g0.w = (unsigned)((gaddr >> 32) & 0x1FFFFFFull) | (2u << 30); // global_addr hi | type=2
  i32x8 g1;
  g1[0] = dataSizeCode << 16;                                 // wg_mask=0, data_size
  g1[1] = (int)((tensorDim0 & 0xFFFFu) << 16);                // tensor_dim0 lo16
  g1[2] = (int)(((tensorDim0 >> 16) & 0xFFFFu) | 0xFFFF0000u);// dim0 hi | tensor_dim1 lo=65535
  g1[3] = (int)((unsigned)tileX << 16);                       // dim1 hi=0 | tile_dim0
  g1[4] = tileY & 0xFFFF;                                     // tile_dim1 | tile_dim2=0
  g1[5] = (int)(strideElems & 0xFFFFFFFFull);                 // tensor_dim0_stride lo32
  g1[6] = (int)((strideElems >> 32) & 0xFFFFull);             // stride hi16 | stride1 lo=0
  g1[7] = 0;
  i32x4 z4 = {0, 0, 0, 0};
#if defined(__clang_major__) && __clang_major__ >= 23
  i32x8 z8 = {0, 0, 0, 0, 0, 0, 0, 0};
  __builtin_amdgcn_tensor_load_to_lds(g0, g1, z4, z4, z8, 0);
#else
  __builtin_amdgcn_tensor_load_to_lds(g0, g1, z4, z4, 0);
#endif
}
#endif

// ---------------- precompute 1/denom [S,S] (L2-resident, reused B*H times) ----
__global__ void denom_kernel(const float* __restrict__ pos, float* __restrict__ dinv) {
  size_t idx = (size_t)blockIdx.x * blockDim.x + threadIdx.x;
  int ii = (int)(idx / S_), jj = (int)(idx % S_);
  const float4* pi = (const float4*)(pos + (size_t)ii * P_);
  const float4* pj = (const float4*)(pos + (size_t)jj * P_);
  float r2 = 0.f;
#pragma unroll
  for (int p = 0; p < P_ / 4; ++p) {
    float4 a = pi[p], b = pj[p];
    float dx = a.x - b.x, dy = a.y - b.y, dz = a.z - b.z, dw = a.w - b.w;
    r2 += dx * dx + dy * dy + dz * dz + dw * dw;
  }
  float den = r2 + 0.15f * r2 * r2 + 1e-6f;
  dinv[idx] = 1.0f / den;
}

// ---------------- fp32 -> bf16 bulk convert (row-major) ----------------
__global__ void cvt_bf16(const float* __restrict__ x, unsigned short* __restrict__ o, int n) {
  int idx = blockIdx.x * blockDim.x + threadIdx.x;
  if (idx < n) o[idx] = f2bf(x[idx]);
}

// ---------------- fp32 W[K,N] -> bf16 Wt[N,K] (one-time, weights are tiny) ----
__global__ void cvt_w_trans(const float* __restrict__ W, unsigned short* __restrict__ Wt,
                            int K, int N) {
  int idx = blockIdx.x * blockDim.x + threadIdx.x;
  if (idx >= K * N) return;
  int k = idx / N, n = idx % N;
  Wt[(size_t)n * K + k] = f2bf(W[idx]);
}

// -------- per-token per-head masses, stored transposed mT[b,h,S] --------------
__global__ void mass_kernel(const float* __restrict__ x, const float* __restrict__ Wm,
                            const float* __restrict__ bm, float* __restrict__ mT) {
  int idx = blockIdx.x * blockDim.x + threadIdx.x;   // [B*S*H]
  int h = idx & (H_ - 1);
  int row = idx >> 3;                                // b*S + s
  const float* xr = x + (size_t)row * D_;
  float acc = bm[h];
  for (int d = 0; d < D_; ++d) acc += xr[d] * Wm[d * H_ + h];
  float m = (acc > 20.0f) ? acc : log1pf(__expf(acc));
  int b = row / S_, s = row - b * S_;
  mT[((size_t)b * H_ + h) * S_ + s] = m;
}

// ------ WMMA GEMM: out = act(A_bf16[M,K] @ Wt_bf16[N,K]^T + bias) -------------
// one wave per 16(M) x 64(N) tile: 4 C fragments, A-frag reused 4x per K-step.
__global__ void wmma_gemm(const unsigned short* __restrict__ A,
                          const unsigned short* __restrict__ Wt,
                          const float* __restrict__ bias, float* __restrict__ outF,
                          unsigned short* __restrict__ outBf,
                          int M, int N, int K, int act, int outBfTrans) {
  int lane = threadIdx.x;
  int i = lane & 15, half = lane >> 4;
  int tm = blockIdx.x, tn = blockIdx.y;
  const unsigned short* arow = A + (size_t)(tm * 16 + i) * K;
  const unsigned short* brow[4];
#pragma unroll
  for (int q = 0; q < 4; ++q)
    brow[q] = Wt + (size_t)(tn * 64 + q * 16 + i) * K;
  v8f c[4] = {{}, {}, {}, {}};
  for (int k0 = 0; k0 < K; k0 += 32) {
    v16u af = load_afrag(arow + k0, half);
#pragma unroll
    for (int q = 0; q < 4; ++q) {
      v16u bf = load_bfrag(brow[q] + k0, half);
      c[q] = wmma_bf16(af, bf, c[q]);
    }
  }
#pragma unroll
  for (int q = 0; q < 4; ++q) {
    int col = tn * 64 + q * 16 + i;
    float bn = bias ? bias[col] : 0.0f;
#pragma unroll
    for (int r = 0; r < 8; ++r) {
      int row = tm * 16 + r + 8 * half;               // C: M = r + 8*half
      float y = c[q][r] + bn;
      if (act == 1) { // gelu (tanh approx, matches jax.nn.gelu default)
        float x3 = y * y * y;
        y = 0.5f * y * (1.0f + tanhf(0.7978845608028654f * (y + 0.044715f * x3)));
      }
      if (outF) outF[(size_t)row * N + col] = y;
      if (outBf) {
        if (outBfTrans) {
          int b = row / S_, s = row - b * S_;
          outBf[((size_t)b * N + col) * S_ + s] = f2bf(y);
        } else {
          outBf[(size_t)row * N + col] = f2bf(y);
        }
      }
    }
  }
}

// ---------------- fused flash gravitational attention ------------------------
// grid: (S/16 i-tiles, H, B), 1 wave/block. Never materializes [B,H,S,S].
// TDM double-buffers the 1/denom (16x32 f32) and V (32x32 bf16) tiles in LDS,
// overlapping the tensor DMA with the softmax VALU work (s_wait_tensorcnt).
__global__ void grav_attn(const float* __restrict__ mT, const float* __restrict__ dinv,
                          const unsigned short* __restrict__ vT, const float* __restrict__ Gh,
                          unsigned short* __restrict__ ctx) {
  int lane = threadIdx.x;
  int i = lane & 15, half = lane >> 4;
  int i0 = blockIdx.x * 16;
  int h = blockIdx.y;
  int b = blockIdx.z;
  const float* mrow = mT + ((size_t)b * H_ + h) * S_;
  float Gmi = Gh[h] * mrow[i0 + i];                   // row-side G_h * m_i
  float rowM = -1e30f, rowL = 0.0f;                   // flash stats for row i (dup in lane+16)
  v8f acc0 = {}, acc1 = {};                           // dh cols 0..15 / 16..31

#if HAVE_TDM
  __shared__ float dtile[2][16 * 32];                 // [i row][j]
  __shared__ unsigned short vtile[2][32 * 32];        // [d col][j]
  const float* dbase = dinv + (size_t)i0 * S_;
  const unsigned short* vbase = vT + ((size_t)b * D_ + h * DH_) * S_;
  // prologue: fill buffer 0
  tdm_load_2d((unsigned)(unsigned long long)&dtile[0][0],
              (unsigned long long)(dbase + 0), 2, 32, 16, S_, S_);
  tdm_load_2d((unsigned)(unsigned long long)&vtile[0][0],
              (unsigned long long)(vbase + 0), 1, 32, 32, S_, S_);
#else
  const float* drow = dinv + (size_t)(i0 + i) * S_;
  const unsigned short* vrow0 = vT + ((size_t)b * D_ + h * DH_ + i) * S_;
  const unsigned short* vrow1 = vrow0 + (size_t)16 * S_;
#endif

  for (int j0 = 0; j0 < S_; j0 += 32) {
#if HAVE_TDM
    int cur = (j0 >> 5) & 1;
    bool has_next = (j0 + 32) < S_;
    if (has_next) {                                   // kick off next pair of tiles
      tdm_load_2d((unsigned)(unsigned long long)&dtile[cur ^ 1][0],
                  (unsigned long long)(dbase + j0 + 32), 2, 32, 16, S_, S_);
      tdm_load_2d((unsigned)(unsigned long long)&vtile[cur ^ 1][0],
                  (unsigned long long)(vbase + j0 + 32), 1, 32, 32, S_, S_);
      __builtin_amdgcn_s_wait_tensorcnt(2);           // current pair landed
    } else {
      __builtin_amdgcn_s_wait_tensorcnt(0);
    }
    const float* dt = &dtile[cur][i * 32];
    float4 dv[4];
    dv[0] = *(const float4*)(dt + 8 * half);
    dv[1] = *(const float4*)(dt + 8 * half + 4);
    dv[2] = *(const float4*)(dt + 16 + 8 * half);
    dv[3] = *(const float4*)(dt + 16 + 8 * half + 4);
#else
    if (j0 + 32 < S_) {
      __builtin_prefetch(drow + j0 + 32, 0, 3);
      __builtin_prefetch(vrow0 + j0 + 32, 0, 3);
    }
    float4 dv[4];
    dv[0] = *(const float4*)(drow + j0 + 8 * half);
    dv[1] = *(const float4*)(drow + j0 + 8 * half + 4);
    dv[2] = *(const float4*)(drow + j0 + 16 + 8 * half);
    dv[3] = *(const float4*)(drow + j0 + 16 + 8 * half + 4);
#endif
    // element order matches A-frag K map: runs [8h..8h+8) then [16+8h..24+8h)
    float4 mv[4];
    mv[0] = *(const float4*)(mrow + j0 + 8 * half);
    mv[1] = *(const float4*)(mrow + j0 + 8 * half + 4);
    mv[2] = *(const float4*)(mrow + j0 + 16 + 8 * half);
    mv[3] = *(const float4*)(mrow + j0 + 16 + 8 * half + 4);
    const float* mj = (const float*)mv;
    const float* dj = (const float*)dv;
    float p[16];
    float lmax = -1e30f;
#pragma unroll
    for (int e = 0; e < 16; ++e) {
      float f = fminf(Gmi * mj[e] * dj[e], FMAX_);
      p[e] = f;
      lmax = fmaxf(lmax, f);
    }
    lmax = fmaxf(lmax, __shfl_xor(lmax, 16, 32));     // combine K halves
    float newM = fmaxf(rowM, lmax);
    float corr = __expf(rowM - newM);
    float psum = 0.0f;
    v16u pu;
#pragma unroll
    for (int e = 0; e < 16; ++e) {
      float pe = __expf(p[e] - newM);
      psum += pe;
      pu[e] = f2bf(pe);
    }
    psum += __shfl_xor(psum, 16, 32);
    rowL = rowL * corr + psum;
    rowM = newM;
#pragma unroll
    for (int r = 0; r < 8; ++r) {                     // rescale accumulators
      float cf = __shfl(corr, r + 8 * half, 32);
      acc0[r] *= cf;
      acc1[r] *= cf;
    }
#if HAVE_TDM
    v16u vb0 = load_bfrag(&vtile[cur][i * 32], half);         // ds_load_b128 x2
    v16u vb1 = load_bfrag(&vtile[cur][(i + 16) * 32], half);
#else
    v16u vb0 = load_bfrag(vrow0 + j0, half);
    v16u vb1 = load_bfrag(vrow1 + j0, half);
#endif
    acc0 = wmma_bf16(pu, vb0, acc0);
    acc1 = wmma_bf16(pu, vb1, acc1);
  }
#pragma unroll
  for (int r = 0; r < 8; ++r) {
    float Lr = __shfl(rowL, r + 8 * half, 32);
    float inv = 1.0f / Lr;
    size_t orow = ((size_t)b * S_ + i0 + r + 8 * half) * D_ + h * DH_ + i;
    ctx[orow]      = f2bf(acc0[r] * inv);             // ctx row-major: A of Wo GEMM
    ctx[orow + 16] = f2bf(acc1[r] * inv);
  }
}

// ---------------- residual + LayerNorm (one block per row of D=256) ----------
__global__ void ln_add_kernel(const float* __restrict__ xa, const float* __restrict__ xb,
                              const float* __restrict__ g, const float* __restrict__ bt,
                              float* __restrict__ outF, unsigned short* __restrict__ outBf) {
  __shared__ float red[D_];
  int row = blockIdx.x, t = threadIdx.x;
  float v = xa[(size_t)row * D_ + t] + xb[(size_t)row * D_ + t];
  red[t] = v;
  __syncthreads();
  for (int s = D_ / 2; s > 0; s >>= 1) { if (t < s) red[t] += red[t + s]; __syncthreads(); }
  float mu = red[0] / D_;
  __syncthreads();
  float dv = v - mu;
  red[t] = dv * dv;
  __syncthreads();
  for (int s = D_ / 2; s > 0; s >>= 1) { if (t < s) red[t] += red[t + s]; __syncthreads(); }
  float var = red[0] / D_;
  float y = dv * rsqrtf(var + LNEPS_) * g[t] + bt[t];
  if (outF)  outF[(size_t)row * D_ + t] = y;
  if (outBf) outBf[(size_t)row * D_ + t] = f2bf(y);
}

extern "C" void kernel_launch(void* const* d_in, const int* in_sizes, int n_in,
                              void* d_out, int out_size, void* d_ws, size_t ws_size,
                              hipStream_t stream) {
  (void)in_sizes; (void)n_in; (void)out_size; (void)ws_size;
  const float* x    = (const float*)d_in[0];
  const float* pos  = (const float*)d_in[1];
  const float* Wm   = (const float*)d_in[2];
  const float* bm   = (const float*)d_in[3];
  const float* Wv   = (const float*)d_in[4];
  const float* bv   = (const float*)d_in[5];
  const float* Wo   = (const float*)d_in[6];
  const float* bo   = (const float*)d_in[7];
  const float* Gh   = (const float*)d_in[8];
  const float* W1   = (const float*)d_in[9];
  const float* b1   = (const float*)d_in[10];
  const float* W2   = (const float*)d_in[11];
  const float* b2   = (const float*)d_in[12];
  const float* ln1g = (const float*)d_in[13];
  const float* ln1b = (const float*)d_in[14];
  const float* ln2g = (const float*)d_in[15];
  const float* ln2b = (const float*)d_in[16];
  float* out = (float*)d_out;

  char* ws = (char*)d_ws;
  size_t off = 0;
  auto alloc = [&](size_t bytes) -> void* {
    void* p = ws + off;
    off += (bytes + 255) & ~(size_t)255;
    return p;
  };
  const size_t M = (size_t)B_ * S_;                                  // 8192 rows
  float*          dinv  = (float*)alloc((size_t)S_ * S_ * 4);        // 16 MB, L2-resident
  unsigned short* xbf   = (unsigned short*)alloc(M * D_ * 2);
  float*          mT    = (float*)alloc(M * H_ * 4);                 // [B,H,S]
  unsigned short* vT    = (unsigned short*)alloc(M * D_ * 2);        // [B,D,S]
  unsigned short* ctxbf = (unsigned short*)alloc(M * D_ * 2);        // [B*S,D]
  float*          attnF = (float*)alloc(M * D_ * 4);
  float*          x1F   = (float*)alloc(M * D_ * 4);
  unsigned short* x1bf  = (unsigned short*)alloc(M * D_ * 2);
  unsigned short* h1bf  = (unsigned short*)alloc(M * FF_ * 2);
  float*          ffnF  = (float*)alloc(M * D_ * 4);
  unsigned short* WvT   = (unsigned short*)alloc((size_t)D_ * D_ * 2);   // [N,K] bf16
  unsigned short* WoT   = (unsigned short*)alloc((size_t)D_ * D_ * 2);
  unsigned short* W1T   = (unsigned short*)alloc((size_t)FF_ * D_ * 2);
  unsigned short* W2T   = (unsigned short*)alloc((size_t)D_ * FF_ * 2);

  // 0) one-time weight convert+transpose to bf16 [N,K]
  cvt_w_trans<<<(D_ * D_) / 256, 256, 0, stream>>>(Wv, WvT, D_, D_);
  cvt_w_trans<<<(D_ * D_) / 256, 256, 0, stream>>>(Wo, WoT, D_, D_);
  cvt_w_trans<<<(D_ * FF_) / 256, 256, 0, stream>>>(W1, W1T, D_, FF_);
  cvt_w_trans<<<(FF_ * D_) / 256, 256, 0, stream>>>(W2, W2T, FF_, D_);
  // 1) 1/denom [S,S]
  denom_kernel<<<(S_ * (size_t)S_) / 256, 256, 0, stream>>>(pos, dinv);
  // 2) x -> bf16 (A matrix of Wv GEMM)
  cvt_bf16<<<(int)(M * D_ / 256), 256, 0, stream>>>(x, xbf, (int)(M * D_));
  // 3) masses (transposed [B,H,S])
  mass_kernel<<<(int)(M * H_ / 256), 256, 0, stream>>>(x, Wm, bm, mT);
  // 4) V = x@Wv + bv, stored transposed vT[b,d,S] for attention B-fragments
  wmma_gemm<<<dim3((int)M / 16, D_ / 64), 32, 0, stream>>>(xbf, WvT, bv, nullptr, vT,
                                                           (int)M, D_, D_, 0, 1);
  // 5) fused flash gravitational attention -> ctx (bf16, row-major)
  grav_attn<<<dim3(S_ / 16, H_, B_), 32, 0, stream>>>(mT, dinv, vT, Gh, ctxbf);
  // 6) attn_out = ctx@Wo + bo (fp32)
  wmma_gemm<<<dim3((int)M / 16, D_ / 64), 32, 0, stream>>>(ctxbf, WoT, bo, attnF, nullptr,
                                                           (int)M, D_, D_, 0, 0);
  // 7) x1 = LN(x + attn_out)  (fp32 + bf16)
  ln_add_kernel<<<(int)M, D_, 0, stream>>>(x, attnF, ln1g, ln1b, x1F, x1bf);
  // 8) h1 = gelu(x1@W1 + b1) (bf16)
  wmma_gemm<<<dim3((int)M / 16, FF_ / 64), 32, 0, stream>>>(x1bf, W1T, b1, nullptr, h1bf,
                                                            (int)M, FF_, D_, 1, 0);
  // 9) ffn = h1@W2 + b2 (fp32)
  wmma_gemm<<<dim3((int)M / 16, D_ / 64), 32, 0, stream>>>(h1bf, W2T, b2, ffnF, nullptr,
                                                           (int)M, D_, FF_, 0, 0);
  // 10) out = LN(x1 + ffn)
  ln_add_kernel<<<(int)M, D_, 0, stream>>>(x1F, ffnF, ln2g, ln2b, out, nullptr);
}
